// AnchorStripeAttention_55946243998378
// MI455X (gfx1250) — matrix operand order repaired and checked by reading, not compiled, not run on hardware
//
#include <hip/hip_runtime.h>
#include <math.h>

typedef __attribute__((ext_vector_type(16))) _Float16 v16h;
typedef __attribute__((ext_vector_type(8)))  float    v8f;

#define LOGIT_MAX 4.605170185988091f

struct FragU { union { v16h v; _Float16 h[16]; unsigned u[8]; }; };

static __device__ inline v8f wmma16(v16h a, v16h b, v8f c) {
  // D = A(16x32 f16) * B(32x16 f16) + C(16x16 f32)
  return __builtin_amdgcn_wmma_f32_16x16x32_f16(false, a, false, b, (short)0, c, false, false);
}

// A fragment: rows [rowBase, rowBase+16), K=0..31, LDS row-major, 32 halves/row.
// Contiguous K order used here: lanes0-15 M=lane; VGPR v: K = (lane>>4)*16 + v*2 (+0,+1)
static __device__ inline v16h loadA32(const _Float16* sm, int rowBase, int lane) {
  FragU f;
  int m = lane & 15, half = lane >> 4;
  const unsigned* p = (const unsigned*)(sm + (rowBase + m) * 32);
#pragma unroll
  for (int v = 0; v < 8; ++v)
    f.u[v] = p[half * 8 + v];
  return f.v;
}

// B fragment, K contiguous per column: B[k][n] = sm[(colBase+n)*32 + k]
// Contiguous K order used here: lane holds column n=lane&15; lanes0-15 K=0..15, lanes16-31 K=16..31
static __device__ inline v16h loadB_kcontig(const _Float16* sm, int colBase, int lane) {
  FragU f;
  int n = lane & 15, half = lane >> 4;
  const unsigned* p = (const unsigned*)(sm + (colBase + n) * 32);
#pragma unroll
  for (int v = 0; v < 8; ++v) f.u[v] = p[half * 8 + v];
  return f.v;
}

// B fragment, K strided: B[k][n] = sm[(kBase+k)*32 + (colBase+n)]
static __device__ inline v16h loadB_kstride(const _Float16* sm, int kBase, int colBase, int lane) {
  FragU f;
  int n = colBase + (lane & 15), half = lane >> 4;
#pragma unroll
  for (int v = 0; v < 8; ++v) {
    int k0 = kBase + half * 16 + 2 * v;
    f.h[2 * v]     = sm[k0 * 32 + n];
    f.h[2 * v + 1] = sm[(k0 + 1) * 32 + n];
  }
  return f.v;
}

static __device__ inline float redmax16(float v) {
#pragma unroll
  for (int m = 1; m < 16; m <<= 1) v = fmaxf(v, __shfl_xor(v, m, 16));
  return v;
}
static __device__ inline float redsum16(float v) {
#pragma unroll
  for (int m = 1; m < 16; m <<= 1) v += __shfl_xor(v, m, 16);
  return v;
}

// CPB MLP: bt[t][h] = (relu(table[t,:] @ w1 + b1) @ w2)[h]   (t < 529, h < 6)
__global__ void cpb_kernel(const float* __restrict__ table, const float* __restrict__ w1,
                           const float* __restrict__ b1, const float* __restrict__ w2,
                           float* __restrict__ bt) {
  int t = blockIdx.x * blockDim.x + threadIdx.x;
  if (t >= 529) return;
  float tx = table[2 * t], ty = table[2 * t + 1];
  float acc[6] = {0.f, 0.f, 0.f, 0.f, 0.f, 0.f};
  for (int j = 0; j < 512; ++j) {
    float hj = fmaxf(tx * w1[j] + ty * w1[512 + j] + b1[j], 0.f);
#pragma unroll
    for (int hh = 0; hh < 6; ++hh) acc[hh] += hj * w2[j * 6 + hh];
  }
#pragma unroll
  for (int hh = 0; hh < 6; ++hh) bt[t * 6 + hh] = acc[hh];
}

// Expand per-(head,pos) bias table: outb[h*N + n] = 16*sigmoid(bt[idx[n]*6 + h])
__global__ void bias_expand(const float* __restrict__ bt, const int* __restrict__ idx,
                            int N, float* __restrict__ outb) {
  int n = blockIdx.x * blockDim.x + threadIdx.x;
  if (n >= N) return;
  int id = idx[n];
#pragma unroll
  for (int hh = 0; hh < 6; ++hh) {
    float x = bt[id * 6 + hh];
    outb[hh * N + n] = 16.f * __builtin_amdgcn_rcpf(1.f + __expf(-x));
  }
}

__global__ __launch_bounds__(128)
void attn_kernel(const float* __restrict__ qkv, const float* __restrict__ anchor,
                 const float* __restrict__ mask_a2w, const float* __restrict__ mask_w2a,
                 const float* __restrict__ ls1, const float* __restrict__ ls2,
                 const float* __restrict__ bias1F, const float* __restrict__ bias2F,
                 float* __restrict__ out) {
  __shared__ __align__(16) _Float16 smK[256 * 32];   // normalized keys
  __shared__ __align__(16) _Float16 smQ[256 * 32];   // normalized queries
  __shared__ __align__(16) _Float16 smV[256 * 32];   // values
  __shared__ __align__(16) _Float16 smAnc[64 * 32];  // normalized anchors
  __shared__ __align__(16) _Float16 smX[64 * 32];    // stage-1 output
  __shared__ __align__(16) _Float16 smP[4 * 16 * 32]; // per-wave P staging

  int bid = blockIdx.x;
  int wglob = bid / 6, h = bid - wglob * 6;     // window id, head
  int b = wglob >> 8, widx = wglob & 255;       // batch, window-in-batch
  int wy = widx >> 4, wx = widx & 15;
  int tid = threadIdx.x;
  int lane = tid & 31, wave = tid >> 5, half = lane >> 4;
  _Float16* pbuf = smP + wave * 512;

  // ---- Phase 0: load + l2norm into LDS (f16) ----
  for (int t = tid; t < 256; t += 128) {
    int i = t >> 4, j = t & 15;
    size_t base = (((size_t)(b * 256 + wy * 16 + i) * 256) + (wx * 16 + j)) * 576 + h * 32;
    {
      float r[32]; float ss = 0.f;
#pragma unroll
      for (int d = 0; d < 32; ++d) { r[d] = qkv[base + d]; ss += r[d] * r[d]; }
      float inv = __builtin_amdgcn_rcpf(fmaxf(sqrtf(ss), 1e-12f));
#pragma unroll
      for (int d = 0; d < 32; ++d) smQ[t * 32 + d] = (_Float16)(r[d] * inv);
    }
    {
      float r[32]; float ss = 0.f;
#pragma unroll
      for (int d = 0; d < 32; ++d) { r[d] = qkv[base + 192 + d]; ss += r[d] * r[d]; }
      float inv = __builtin_amdgcn_rcpf(fmaxf(sqrtf(ss), 1e-12f));
#pragma unroll
      for (int d = 0; d < 32; ++d) smK[t * 32 + d] = (_Float16)(r[d] * inv);
    }
#pragma unroll
    for (int d = 0; d < 32; ++d) smV[t * 32 + d] = (_Float16)qkv[base + 384 + d];
  }
  for (int t = tid; t < 64; t += 128) {
    int i = t >> 3, j = t & 7;
    size_t base = (((size_t)(b * 128 + wy * 8 + i) * 128) + (wx * 8 + j)) * 192 + h * 32;
    float r[32]; float ss = 0.f;
#pragma unroll
    for (int d = 0; d < 32; ++d) { r[d] = anchor[base + d]; ss += r[d] * r[d]; }
    float inv = __builtin_amdgcn_rcpf(fmaxf(sqrtf(ss), 1e-12f));
#pragma unroll
    for (int d = 0; d < 32; ++d) smAnc[t * 32 + d] = (_Float16)(r[d] * inv);
  }
  __syncthreads();

  // ---- Stage 1: anchor -> window. Wave owns anchor rows [16*wave, 16*wave+16) ----
  {
    float scale1 = __expf(fminf(ls1[h], LOGIT_MAX));
    v16h aAnc = loadA32(smAnc, wave * 16, lane);
    float s[16][8];
#pragma unroll
    for (int nt = 0; nt < 16; ++nt) {
      v16h bK = loadB_kcontig(smK, nt * 16, lane);
      v8f c = {};
      c = wmma16(aAnc, bK, c);
      int ncol = nt * 16 + (lane & 15);
#pragma unroll
      for (int r = 0; r < 8; ++r) {
        int m = wave * 16 + r + half * 8;                        // anchor row n2
        float bias = bias1F[(h * 64 + m) * 256 + ncol];
        float mk = mask_a2w[((size_t)widx * 64 + m) * 256 + ncol];
        s[nt][r] = c[r] * scale1 + bias + mk;
      }
    }
    float rs[8];
#pragma unroll
    for (int r = 0; r < 8; ++r) {
      float mx = -1e30f;
#pragma unroll
      for (int nt = 0; nt < 16; ++nt) mx = fmaxf(mx, s[nt][r]);
      mx = redmax16(mx);
      float sum = 0.f;
#pragma unroll
      for (int nt = 0; nt < 16; ++nt) { s[nt][r] = __expf(s[nt][r] - mx); sum += s[nt][r]; }
      rs[r] = redsum16(sum);   // lane-aligned with C/D row (r + half*8)
    }
    v8f xa0 = {}, xa1 = {};
#pragma unroll
    for (int ch = 0; ch < 8; ++ch) {     // K chunks of 32 tokens
#pragma unroll
      for (int t2 = 0; t2 < 2; ++t2)
#pragma unroll
        for (int r = 0; r < 8; ++r)
          pbuf[(r + half * 8) * 32 + t2 * 16 + (lane & 15)] = (_Float16)s[2 * ch + t2][r];
      v16h aP = loadA32(pbuf, 0, lane);  // same-wave LDS ops stay in order
      v16h b0 = loadB_kstride(smV, 32 * ch, 0, lane);
      xa0 = wmma16(aP, b0, xa0);
      v16h b1 = loadB_kstride(smV, 32 * ch, 16, lane);
      xa1 = wmma16(aP, b1, xa1);
    }
#pragma unroll
    for (int r = 0; r < 8; ++r) {
      float inv = __builtin_amdgcn_rcpf(rs[r]);
      int row = wave * 16 + r + half * 8;
      smX[row * 32 + (lane & 15)]      = (_Float16)(xa0[r] * inv);
      smX[row * 32 + 16 + (lane & 15)] = (_Float16)(xa1[r] * inv);
    }
  }
  __syncthreads();

  // ---- Stage 2: window -> anchor. Wave owns q rows [64*wave, 64*wave+64) ----
  float scale2 = __expf(fminf(ls2[h], LOGIT_MAX));
  for (int sb = 0; sb < 4; ++sb) {
    int rbase = wave * 64 + sb * 16;
    v16h aQ = loadA32(smQ, rbase, lane);
    float s2[4][8];
#pragma unroll
    for (int nt = 0; nt < 4; ++nt) {
      v16h bA = loadB_kcontig(smAnc, nt * 16, lane);
      v8f c = {};
      c = wmma16(aQ, bA, c);
      int ncol = nt * 16 + (lane & 15);
#pragma unroll
      for (int r = 0; r < 8; ++r) {
        int n1 = rbase + r + half * 8;
        float bias = bias2F[(h * 256 + n1) * 64 + ncol];
        float mk = mask_w2a[((size_t)widx * 256 + n1) * 64 + ncol];
        s2[nt][r] = c[r] * scale2 + bias + mk;
      }
    }
    float rs2[8];
#pragma unroll
    for (int r = 0; r < 8; ++r) {
      float mx = -1e30f;
#pragma unroll
      for (int nt = 0; nt < 4; ++nt) mx = fmaxf(mx, s2[nt][r]);
      mx = redmax16(mx);
      float sum = 0.f;
#pragma unroll
      for (int nt = 0; nt < 4; ++nt) { s2[nt][r] = __expf(s2[nt][r] - mx); sum += s2[nt][r]; }
      rs2[r] = redsum16(sum);
    }
    v8f oa0 = {}, oa1 = {};
#pragma unroll
    for (int ch = 0; ch < 2; ++ch) {     // K chunks of 32 anchors
#pragma unroll
      for (int t2 = 0; t2 < 2; ++t2)
#pragma unroll
        for (int r = 0; r < 8; ++r)
          pbuf[(r + half * 8) * 32 + t2 * 16 + (lane & 15)] = (_Float16)s2[2 * ch + t2][r];
      v16h aP = loadA32(pbuf, 0, lane);
      v16h b0 = loadB_kstride(smX, 32 * ch, 0, lane);
      oa0 = wmma16(aP, b0, oa0);
      v16h b1 = loadB_kstride(smX, 32 * ch, 16, lane);
      oa1 = wmma16(aP, b1, oa1);
    }
#pragma unroll
    for (int r = 0; r < 8; ++r) {
      float inv = __builtin_amdgcn_rcpf(rs2[r]);
      int n1 = rbase + r + half * 8;
      int grow = wy * 16 + (n1 >> 4), gcol = wx * 16 + (n1 & 15);
      size_t ob = (((size_t)(b * 256 + grow) * 256) + gcol) * 192 + h * 32;
      out[ob + (lane & 15)]      = oa0[r] * inv;
      out[ob + 16 + (lane & 15)] = oa1[r] * inv;
    }
  }
}

extern "C" void kernel_launch(void* const* d_in, const int* in_sizes, int n_in,
                              void* d_out, int out_size, void* d_ws, size_t ws_size,
                              hipStream_t stream) {
  const float* qkv      = (const float*)d_in[0];
  const float* anchor   = (const float*)d_in[1];
  const float* table    = (const float*)d_in[2];
  const float* mask_a2w = (const float*)d_in[3];
  const float* mask_w2a = (const float*)d_in[4];
  const float* ls1      = (const float*)d_in[5];
  const float* w1a      = (const float*)d_in[6];
  const float* b1a      = (const float*)d_in[7];
  const float* w2a      = (const float*)d_in[8];
  const float* ls2      = (const float*)d_in[9];
  const float* w1b      = (const float*)d_in[10];
  const float* b1b      = (const float*)d_in[11];
  const float* w2b      = (const float*)d_in[12];
  const int*   idx_a2w  = (const int*)d_in[13];
  const int*   idx_w2a  = (const int*)d_in[14];
  float* out = (float*)d_out;

  // workspace layout (floats): bt1 @0, bt2 @4096, bias1F @8192 (6*64*256),
  // bias2F @8192+98304 (6*256*64). Total ~819KB.
  float* bt1    = (float*)d_ws;
  float* bt2    = bt1 + 4096;
  float* bias1F = bt1 + 8192;
  float* bias2F = bias1F + 6 * 64 * 256;

  cpb_kernel<<<dim3((529 + 63) / 64), dim3(64), 0, stream>>>(table, w1a, b1a, w2a, bt1);
  cpb_kernel<<<dim3((529 + 63) / 64), dim3(64), 0, stream>>>(table, w1b, b1b, w2b, bt2);
  bias_expand<<<dim3((16384 + 127) / 128), dim3(128), 0, stream>>>(bt1, idx_a2w, 16384, bias1F);
  bias_expand<<<dim3((16384 + 127) / 128), dim3(128), 0, stream>>>(bt2, idx_w2a, 16384, bias2F);
  attn_kernel<<<dim3(3072), dim3(128), 0, stream>>>(qkv, anchor, mask_a2w, mask_w2a,
                                                    ls1, ls2, bias1F, bias2F, out);
}
